// Network_48747878810290
// MI455X (gfx1250) — compile-verified, matrix operand order checked by reading, not run on hardware
//
#include <hip/hip_runtime.h>

typedef __attribute__((ext_vector_type(2)))  float    v2f;
typedef __attribute__((ext_vector_type(8)))  float    v8f;
typedef __attribute__((ext_vector_type(16))) _Float16 v16h;

// One Izhikevich neuron update, TAU*DT = 250*0.001 = 0.25 baked in.
__device__ __forceinline__ float izh_step(float& v, float& u, float I,
                                          float a, float b, float c, float d)
{
    float v1 = v + 0.25f * (0.04f * v * v + 5.0f * v + 140.0f - u + I);
    float u1 = u + 0.25f * a * (b * v - u);
    float z  = (v1 >= 30.0f) ? 1.0f : 0.0f;
    v = (z > 0.0f) ? c : v1;
    u = u1 + z * d;
    return z;
}

__global__ __launch_bounds__(32) void izh_net_kernel(
    const float* __restrict__ A,   // input_mat, N elements
    const float* __restrict__ w,   // 12 weights
    float* __restrict__ out,       // [5][T] spikes then [5][T] volts
    int N, int T)
{
    __shared__ float red[32];
    const int lane = threadIdx.x;

    float partial = 0.0f;

    // ---- Stage 1: total = sum(A) via WMMA with an all-ones B matrix ----
    // D[m][n] = rowsum_m(A)  =>  sum over all 256 D elements = 16 * total.
#if __has_builtin(__builtin_amdgcn_wmma_f32_16x16x4_f32)
    {
        // Exact f32 path: V_WMMA_F32_16X16X4_F32, 64 f32 elements per op.
        v8f c = {};
        v2f bones; bones[0] = 1.0f; bones[1] = 1.0f;
        const int chunks = N >> 6;
        for (int ch = 0; ch < chunks; ++ch) {
            const int base = (ch << 6) + (lane << 1);
            v2f a; a[0] = A[base]; a[1] = A[base + 1];
            c = __builtin_amdgcn_wmma_f32_16x16x4_f32(false, a, false, bones,
                                                      (short)0, c, false, false);
        }
        partial = (c[0]+c[1]+c[2]+c[3]+c[4]+c[5]+c[6]+c[7]) * 0.0625f;
        if (lane == 0)
            for (int i = chunks << 6; i < N; ++i) partial += A[i];
    }
#else
    {
        // Fallback: probe-confirmed V_WMMA_F32_16X16X32_F16, 512 elems per op.
        v8f c = {};
        v16h bones;
        #pragma unroll
        for (int i = 0; i < 16; ++i) bones[i] = (_Float16)1.0f;
        const int chunks = N >> 9;
        for (int ch = 0; ch < chunks; ++ch) {
            const int base = (ch << 9) + (lane << 4);
            v16h a;
            #pragma unroll
            for (int i = 0; i < 16; ++i) a[i] = (_Float16)A[base + i];
            c = __builtin_amdgcn_wmma_f32_16x16x32_f16(false, a, false, bones,
                                                       (short)0, c, false, false);
        }
        partial = (c[0]+c[1]+c[2]+c[3]+c[4]+c[5]+c[6]+c[7]) * 0.0625f;
        if (lane == 0)
            for (int i = chunks << 9; i < N; ++i) partial += A[i];
    }
#endif

    red[lane] = partial;
    __syncthreads();
    if (lane != 0) return;

    float S = 0.0f;
    #pragma unroll
    for (int i = 0; i < 32; ++i) S += red[i];

    // ---- Stage 2: serial 5-neuron Izhikevich recurrence, one lane ----
    const float w0 = w[0], w1 = w[1], w2 = w[2], w3 = w[3], w4 = w[4],
                w5 = w[5], w6 = w[6], w7 = w[7], w8 = w[8], w9 = w[9],
                w10 = w[10], w11 = w[11];
    const float zplus = S * w0;

    float temp = 0.0f;
    float z2p = 0.0f, v2 = -60.0f, u2 =   4.5f;   // LLBN (u0 = b*v_rest)
    float             v3 = -64.0f, u3 = -16.0f;   // EBN
    float             v4 = -64.0f, u4 = -16.0f;   // IFN
    float z5p = 0.0f, v5 = -70.0f, u5 = -14.0f;   // TN
    float             v6 = -64.0f, u6 = -16.0f;   // MN

    float* sp = out;          // spikes [5][T]
    float* vo = out + 5 * T;  // volts  [5][T]

    for (int t = 0; t < T; ++t) {
        float z2 = izh_step(v2, u2, w2 * (zplus * w1 - temp) + w3 * z2p,
                            0.1f, -0.075f, -55.0f, 6.0f);          // LLBN
        float z3 = izh_step(v3, u3, zplus * w4 + z2 * w5,
                            0.02f, 0.25f, -55.0f, 0.05f);          // EBN
        float z4 = izh_step(v4, u4, z3 * w6,
                            0.02f, 0.25f, -65.0f, 6.0f);           // IFN
        temp = z4 * w7;                                            // 1-step delay fb
        float z5 = izh_step(v5, u5, w9 * (z3 * w8) + w10 * z5p,
                            0.02f, 0.2f, -65.0f, 6.0f);            // TN
        float z6 = izh_step(v6, u6, z5 * w11,
                            0.02f, 0.25f, -65.0f, 6.0f);           // MN
        z2p = z2; z5p = z5;

        sp[t]         = z2;
        sp[T + t]     = z3;
        sp[2 * T + t] = z4;
        sp[3 * T + t] = z5;
        sp[4 * T + t] = z6;
        vo[t]         = v2;
        vo[T + t]     = v3;
        vo[2 * T + t] = v4;
        vo[3 * T + t] = v5;
        vo[4 * T + t] = v6;
    }
}

extern "C" void kernel_launch(void* const* d_in, const int* in_sizes, int n_in,
                              void* d_out, int out_size, void* d_ws, size_t ws_size,
                              hipStream_t stream)
{
    const float* A = (const float*)d_in[0];   // input_mat (127*127 f32)
    const float* w = (const float*)d_in[1];   // weights (12 f32)
    const int N = in_sizes[0];
    const int T = out_size / 10;              // out = 5*T spikes + 5*T volts
    izh_net_kernel<<<dim3(1), dim3(32), 0, stream>>>(A, w, (float*)d_out, N, T);
}
